// TransformerDecoderLayer_20839181320782
// MI455X (gfx1250) — compile-verified, matrix operand order checked by reading, not run on hardware
//
#include <hip/hip_runtime.h>
#include <math.h>
#include <stdint.h>

// ---------------- constants ----------------
static constexpr int CB  = 4;      // batch
static constexpr int CN  = 1024;   // seq len
static constexpr int CD  = 1024;   // model dim
static constexpr int CH  = 16;     // heads
static constexpr int CDH = 64;     // head dim
static constexpr int CDM = 4096;   // mlp dim
static constexpr float CSCALE = 0.125f;  // 64^-0.5

typedef float  v8f   __attribute__((ext_vector_type(8)));
typedef __bf16 v16bf __attribute__((ext_vector_type(16)));
typedef __bf16 bf8v  __attribute__((ext_vector_type(8)));

static __device__ __forceinline__ v8f wmma_bf16(v16bf a, v16bf b, v8f c) {
  return __builtin_amdgcn_wmma_f32_16x16x32_bf16(false, a, false, b, (short)0, c, false, false);
}

// load one 16x32 bf16 operand fragment slice for this lane:
// 8 consecutive bf16 at p and 8 more at p+16 (ISA 16-bit A/B layout).
static __device__ __forceinline__ v16bf frag_ld(const __bf16* p) {
  bf8v lo = *(const bf8v*)(p);
  bf8v hi = *(const bf8v*)(p + 16);
  return __builtin_shufflevector(lo, hi, 0,1,2,3,4,5,6,7,8,9,10,11,12,13,14,15);
}

static __device__ __forceinline__ float gelu_exact(float x) {
  return 0.5f * x * (1.0f + erff(x * 0.70710678118654752f));
}

// issue one async memory->LDS 16B copy per lane (gfx1250, tracked by ASYNCcnt)
static __device__ __forceinline__ void async_ld_b128(unsigned lds_byte_addr,
                                                     unsigned voff_bytes,
                                                     unsigned long long sbase) {
  asm volatile("global_load_async_to_lds_b128 %0, %1, %2"
               :: "v"(lds_byte_addr), "v"(voff_bytes), "s"(sbase)
               : "memory");
}
static __device__ __forceinline__ void wait_asynccnt0() {
  asm volatile("s_wait_asynccnt 0x0" ::: "memory");
}

// ---------------- weight transpose + f32->bf16 convert ----------------
// W is K x N (row-major, "in x out"); Wt is N x K bf16.
__global__ __launch_bounds__(256) void wconv_kernel(const float* __restrict__ W,
                                                    __bf16* __restrict__ Wt,
                                                    int K, int N) {
  __shared__ float t[32][33];
  int tx = threadIdx.x, ty = threadIdx.y;
  int n0 = blockIdx.x * 32, k0 = blockIdx.y * 32;
#pragma unroll
  for (int yy = 0; yy < 4; ++yy) {
    int kk = ty * 4 + yy;
    t[kk][tx] = W[(size_t)(k0 + kk) * N + (n0 + tx)];
  }
  __syncthreads();
#pragma unroll
  for (int yy = 0; yy < 4; ++yy) {
    int nn = ty * 4 + yy;
    Wt[(size_t)(n0 + nn) * K + (k0 + tx)] = (__bf16)t[tx][nn];
  }
}

// ---------------- element-wise helpers ----------------
__global__ __launch_bounds__(256) void copy_kernel(const float* __restrict__ a,
                                                   float* __restrict__ o, int n4) {
  int i = blockIdx.x * blockDim.x + threadIdx.x;
  if (i < n4) ((float4*)o)[i] = ((const float4*)a)[i];
}

__global__ __launch_bounds__(256) void addpe_kernel(const float* __restrict__ a,
                                                    const float* __restrict__ b,
                                                    float* __restrict__ o, int n4) {
  int i = blockIdx.x * blockDim.x + threadIdx.x;
  if (i < n4) {
    float4 x = ((const float4*)a)[i];
    float4 y = ((const float4*)b)[i];
    float4 r{ x.x + y.x, x.y + y.y, x.z + y.z, x.w + y.w };
    ((float4*)o)[i] = r;
  }
}

// ---------------- LayerNorm (D = 1024, one row per block, 256 threads) ----------
__global__ __launch_bounds__(256) void ln_kernel(const float* __restrict__ x,
                                                 const float* __restrict__ g,
                                                 const float* __restrict__ bt,
                                                 const float* __restrict__ pe,
                                                 float* __restrict__ x2,
                                                 float* __restrict__ qk) {
  __shared__ float s1[256], s2[256];
  int tid = threadIdx.x;
  size_t base = (size_t)blockIdx.x * CD;
  float4 f = *(const float4*)(x + base + tid * 4);
  float sum = f.x + f.y + f.z + f.w;
  float sq  = f.x*f.x + f.y*f.y + f.z*f.z + f.w*f.w;
  s1[tid] = sum; s2[tid] = sq;
  __syncthreads();
  for (int s = 128; s > 0; s >>= 1) {
    if (tid < s) { s1[tid] += s1[tid + s]; s2[tid] += s2[tid + s]; }
    __syncthreads();
  }
  float mu  = s1[0] * (1.0f / CD);
  float var = s2[0] * (1.0f / CD) - mu * mu;
  float rstd = rsqrtf(var + 1e-5f);
  float4 g4 = *(const float4*)(g  + tid * 4);
  float4 b4 = *(const float4*)(bt + tid * 4);
  float4 y;
  y.x = (f.x - mu) * rstd * g4.x + b4.x;
  y.y = (f.y - mu) * rstd * g4.y + b4.y;
  y.z = (f.z - mu) * rstd * g4.z + b4.z;
  y.w = (f.w - mu) * rstd * g4.w + b4.w;
  *(float4*)(x2 + base + tid * 4) = y;
  if (qk) {
    float4 p = *(const float4*)(pe + base + tid * 4);
    float4 q{ y.x + p.x, y.y + p.y, y.z + p.z, y.w + p.w };
    *(float4*)(qk + base + tid * 4) = q;
  }
}

// ---------------- generic bf16 WMMA GEMM (double-buffered, async B stage) ------
// C[M,N] = act( A[M,K](f32) @ Bt[N,K](bf16)^T + bias[N] ) (+ resid[M,N])
// flags: bit0 = GELU, bit1 = residual add
#define GBM 128
#define GBN 128
#define GBK 64
#define GLD 72   // padded LDS row stride (bf16 elems): 144B, bank-conflict free

__global__ __launch_bounds__(256) void gemm_kernel(const float*  __restrict__ A,
                                                   const __bf16* __restrict__ Bt,
                                                   const float*  __restrict__ bias,
                                                   const float*  __restrict__ resid,
                                                   float* __restrict__ C,
                                                   int M, int N, int K, int flags) {
  __shared__ __align__(16) __bf16 As[2][GBM * GLD];
  __shared__ __align__(16) __bf16 Bs[2][GBN * GLD];

  int tid = threadIdx.x;
  int lane = tid & 31, w = tid >> 5;
  int wm = w >> 2, wn = w & 3;          // 2x4 wave grid -> 64x32 per wave
  int hlf = lane >> 4, l16 = lane & 15;
  int bm = blockIdx.y * GBM, bn = blockIdx.x * GBN;

  v8f acc[4][2];
#pragma unroll
  for (int i = 0; i < 4; ++i)
#pragma unroll
    for (int j = 0; j < 2; ++j) acc[i][j] = (v8f)(0.0f);

  int ar = tid >> 4, ac = (tid & 15) * 4;   // A stage coords (16 rows x 16 float4)
  int br = tid >> 3, bc = (tid & 7) * 8;    // B stage coords (32 rows x 8 bf8)

  // per-lane byte offsets into Bt tile rows (constant across k-tiles)
  unsigned voffB[4];
#pragma unroll
  for (int rr = 0; rr < 4; ++rr)
    voffB[rr] = (unsigned)((((size_t)(br + rr * 32)) * K + bc) * 2);

  // ---- preload tile 0 ----
  {
    unsigned long long sb = (unsigned long long)(Bt + (size_t)bn * K);
#pragma unroll
    for (int rr = 0; rr < 4; ++rr) {
      unsigned ldsb = (unsigned)(uintptr_t)&Bs[0][(br + rr * 32) * GLD + bc];
      async_ld_b128(ldsb, voffB[rr], sb);
    }
#pragma unroll
    for (int rr = 0; rr < 8; ++rr) {
      int row = ar + rr * 16;
      float4 f = *(const float4*)(A + (size_t)(bm + row) * K + ac);
      __bf16* d = &As[0][row * GLD + ac];
      d[0] = (__bf16)f.x; d[1] = (__bf16)f.y; d[2] = (__bf16)f.z; d[3] = (__bf16)f.w;
    }
    wait_asynccnt0();
  }
  __syncthreads();

  int buf = 0;
  for (int k0 = 0; k0 < K; k0 += GBK) {
    int nk = k0 + GBK;
    bool more = nk < K;
    float4 pf[8];
    if (more) {
      // issue next A-tile global loads (held in regs, overlap with compute)
#pragma unroll
      for (int rr = 0; rr < 8; ++rr)
        pf[rr] = *(const float4*)(A + (size_t)(bm + ar + rr * 16) * K + nk + ac);
      // stream next B-tile straight into the other LDS buffer (ASYNCcnt)
      unsigned long long sb = (unsigned long long)(Bt + (size_t)bn * K + nk);
#pragma unroll
      for (int rr = 0; rr < 4; ++rr) {
        unsigned ldsb = (unsigned)(uintptr_t)&Bs[buf ^ 1][(br + rr * 32) * GLD + bc];
        async_ld_b128(ldsb, voffB[rr], sb);
      }
    }

    // ---- compute on current buffer ----
#pragma unroll
    for (int ks = 0; ks < GBK; ks += 32) {
      int kb = ks + hlf * 8;
      v16bf af[4], bv[2];
#pragma unroll
      for (int i = 0; i < 4; ++i)
        af[i] = frag_ld(&As[buf][(wm * 64 + i * 16 + l16) * GLD + kb]);
#pragma unroll
      for (int j = 0; j < 2; ++j)
        bv[j] = frag_ld(&Bs[buf][(wn * 32 + j * 16 + l16) * GLD + kb]);
#pragma unroll
      for (int i = 0; i < 4; ++i)
#pragma unroll
        for (int j = 0; j < 2; ++j)
          acc[i][j] = wmma_bf16(af[i], bv[j], acc[i][j]);
    }

    if (more) {
      // convert + store next A tile, then make async B tile visible
#pragma unroll
      for (int rr = 0; rr < 8; ++rr) {
        int row = ar + rr * 16;
        __bf16* d = &As[buf ^ 1][row * GLD + ac];
        d[0] = (__bf16)pf[rr].x; d[1] = (__bf16)pf[rr].y;
        d[2] = (__bf16)pf[rr].z; d[3] = (__bf16)pf[rr].w;
      }
      wait_asynccnt0();
    }
    __syncthreads();
    buf ^= 1;
  }

  // ---- epilogue ----
#pragma unroll
  for (int j = 0; j < 2; ++j) {
    int n = bn + wn * 32 + j * 16 + l16;
    float bvv = bias[n];
#pragma unroll
    for (int i = 0; i < 4; ++i) {
      int mrow = bm + wm * 64 + i * 16 + hlf * 8;
#pragma unroll
      for (int r = 0; r < 8; ++r) {
        float v = acc[i][j][r] + bvv;
        if (flags & 1) v = gelu_exact(v);
        size_t idx = (size_t)(mrow + r) * N + n;
        if (flags & 2) v += resid[idx];
        C[idx] = v;
      }
    }
  }
}

// ---------------- flash attention (bf16 WMMA, online softmax) ----------------
// q,k,v,o are (B*N, D) f32 row-major; head h occupies cols [h*64, h*64+64).
#define ABQ 128
#define ABK 128
#define ALQ 72    // Q/K LDS stride (DH=64 + 8 pad)
#define ALP 136   // P/Vt LDS stride (128 + 8 pad)

__global__ __launch_bounds__(256) void attn_kernel(const float* __restrict__ q,
                                                   const float* __restrict__ k,
                                                   const float* __restrict__ v,
                                                   float* __restrict__ o) {
  __shared__ __align__(16) __bf16 Qs[ABQ * ALQ];
  __shared__ __align__(16) __bf16 Ks[ABK * ALQ];
  __shared__ __align__(16) __bf16 Vt[CDH * ALP];
  __shared__ __align__(16) __bf16 Pb[ABQ * ALP];
  __shared__ float red[4][ABQ];
  __shared__ float mstate[ABQ], lstate[ABQ], fac[ABQ];

  int tid = threadIdx.x;
  int lane = tid & 31, w = tid >> 5;
  int wm = w >> 2, wn = w & 3;
  int hlf = lane >> 4, l16 = lane & 15;

  int bh = blockIdx.y;
  int b = bh / CH, h = bh % CH;
  int q0 = blockIdx.x * ABQ;
  size_t headoff = (size_t)h * CDH;
  const float* qb = q + (size_t)b * CN * CD + headoff;
  const float* kb = k + (size_t)b * CN * CD + headoff;
  const float* vb = v + (size_t)b * CN * CD + headoff;

  if (tid < ABQ) { mstate[tid] = -1e30f; lstate[tid] = 0.0f; }

  // stage Q tile (scale folded in)
  {
    int ar = tid >> 4, ac = (tid & 15) * 4;
#pragma unroll
    for (int rr = 0; rr < 8; ++rr) {
      int row = ar + rr * 16;
      float4 f = *(const float4*)(qb + (size_t)(q0 + row) * CD + ac);
      __bf16* d = &Qs[row * ALQ + ac];
      d[0] = (__bf16)(f.x * CSCALE); d[1] = (__bf16)(f.y * CSCALE);
      d[2] = (__bf16)(f.z * CSCALE); d[3] = (__bf16)(f.w * CSCALE);
    }
  }

  v8f oacc[4];
#pragma unroll
  for (int i = 0; i < 4; ++i) oacc[i] = (v8f)(0.0f);

  for (int kt = 0; kt < CN / ABK; ++kt) {
    __syncthreads();  // protect Ks/Vt/Pb reuse (and initial m/l init)
    int kvbase = kt * ABK;
    {
      int ar = tid >> 4, ac = (tid & 15) * 4;
#pragma unroll
      for (int rr = 0; rr < 8; ++rr) {
        int row = ar + rr * 16;
        float4 f = *(const float4*)(kb + (size_t)(kvbase + row) * CD + ac);
        __bf16* d = &Ks[row * ALQ + ac];
        d[0] = (__bf16)f.x; d[1] = (__bf16)f.y; d[2] = (__bf16)f.z; d[3] = (__bf16)f.w;
      }
      int d4 = (tid & 15) * 4, kk0 = tid >> 4;
#pragma unroll
      for (int it = 0; it < 8; ++it) {
        int kk = kk0 + it * 16;
        float4 f = *(const float4*)(vb + (size_t)(kvbase + kk) * CD + d4);
        Vt[(d4 + 0) * ALP + kk] = (__bf16)f.x;
        Vt[(d4 + 1) * ALP + kk] = (__bf16)f.y;
        Vt[(d4 + 2) * ALP + kk] = (__bf16)f.z;
        Vt[(d4 + 3) * ALP + kk] = (__bf16)f.w;
      }
    }
    __syncthreads();

    // S = Qs (128x64) x Ks^T (64x128), wave tile 64x32
    v8f sacc[4][2];
#pragma unroll
    for (int i = 0; i < 4; ++i)
#pragma unroll
      for (int j = 0; j < 2; ++j) sacc[i][j] = (v8f)(0.0f);
#pragma unroll
    for (int ks = 0; ks < CDH; ks += 32) {
      int kbi = ks + hlf * 8;
      v16bf af[4], bvv[2];
#pragma unroll
      for (int i = 0; i < 4; ++i)
        af[i] = frag_ld(&Qs[(wm * 64 + i * 16 + l16) * ALQ + kbi]);
#pragma unroll
      for (int j = 0; j < 2; ++j)
        bvv[j] = frag_ld(&Ks[(wn * 32 + j * 16 + l16) * ALQ + kbi]);
#pragma unroll
      for (int i = 0; i < 4; ++i)
#pragma unroll
        for (int j = 0; j < 2; ++j)
          sacc[i][j] = wmma_bf16(af[i], bvv[j], sacc[i][j]);
    }

    // per-row max over this wave's 32 columns
#pragma unroll
    for (int i = 0; i < 4; ++i) {
      float rm[8];
#pragma unroll
      for (int r = 0; r < 8; ++r) rm[r] = fmaxf(sacc[i][0][r], sacc[i][1][r]);
#pragma unroll
      for (int mask = 1; mask < 16; mask <<= 1)
#pragma unroll
        for (int r = 0; r < 8; ++r) rm[r] = fmaxf(rm[r], __shfl_xor(rm[r], mask, 32));
      if (l16 == 0) {
#pragma unroll
        for (int r = 0; r < 8; ++r)
          red[wn][wm * 64 + i * 16 + hlf * 8 + r] = rm[r];
      }
    }
    __syncthreads();
    if (tid < ABQ) {
      float nm = fmaxf(fmaxf(red[0][tid], red[1][tid]), fmaxf(red[2][tid], red[3][tid]));
      float mold = mstate[tid];
      float mnew = fmaxf(mold, nm);
      fac[tid] = __expf(mold - mnew);
      mstate[tid] = mnew;
    }
    __syncthreads();

    // probabilities -> Pb (bf16), local row sums, rescale O
#pragma unroll
    for (int i = 0; i < 4; ++i) {
      float sm[8];
#pragma unroll
      for (int r = 0; r < 8; ++r) sm[r] = 0.0f;
#pragma unroll
      for (int j = 0; j < 2; ++j) {
        int col = wn * 32 + j * 16 + l16;
#pragma unroll
        for (int r = 0; r < 8; ++r) {
          int row = wm * 64 + i * 16 + hlf * 8 + r;
          float p = __expf(sacc[i][j][r] - mstate[row]);
          Pb[row * ALP + col] = (__bf16)p;
          sm[r] += p;
        }
      }
#pragma unroll
      for (int mask = 1; mask < 16; mask <<= 1)
#pragma unroll
        for (int r = 0; r < 8; ++r) sm[r] += __shfl_xor(sm[r], mask, 32);
      if (l16 == 0) {
#pragma unroll
        for (int r = 0; r < 8; ++r)
          red[wn][wm * 64 + i * 16 + hlf * 8 + r] = sm[r];
      }
#pragma unroll
      for (int r = 0; r < 8; ++r) {
        int row = wm * 64 + i * 16 + hlf * 8 + r;
        oacc[i][r] *= fac[row];
      }
    }
    __syncthreads();
    if (tid < ABQ) {
      float s = red[0][tid] + red[1][tid] + red[2][tid] + red[3][tid];
      lstate[tid] = lstate[tid] * fac[tid] + s;
    }
    __syncthreads();

    // O += P (128x128) x V (128x64); wave tile 64x16
#pragma unroll
    for (int ks = 0; ks < ABK; ks += 32) {
      int kbi = ks + hlf * 8;
      v16bf vf = frag_ld(&Vt[(wn * 16 + l16) * ALP + kbi]);
#pragma unroll
      for (int i = 0; i < 4; ++i) {
        v16bf pf = frag_ld(&Pb[(wm * 64 + i * 16 + l16) * ALP + kbi]);
        oacc[i] = wmma_bf16(pf, vf, oacc[i]);
      }
    }
  }
  __syncthreads();

  // normalize + write out
  float* ob = o + (size_t)b * CN * CD + headoff;
#pragma unroll
  for (int i = 0; i < 4; ++i) {
#pragma unroll
    for (int r = 0; r < 8; ++r) {
      int row = wm * 64 + i * 16 + hlf * 8 + r;
      float vv = oacc[i][r] / lstate[row];
      ob[(size_t)(q0 + row) * CD + wn * 16 + l16] = vv;
    }
  }
}

// ---------------- host launcher ----------------
extern "C" void kernel_launch(void* const* d_in, const int* in_sizes, int n_in,
                              void* d_out, int out_size, void* d_ws, size_t ws_size,
                              hipStream_t stream) {
  const float* tokens_enc = (const float*)d_in[0];
  const float* tokens_dec = (const float*)d_in[1];
  const float* PE_enc     = (const float*)d_in[2];
  const float* PE_dec     = (const float*)d_in[3];
  const float* sa_Wq = (const float*)d_in[4];  const float* sa_bq = (const float*)d_in[5];
  const float* sa_Wk = (const float*)d_in[6];  const float* sa_bk = (const float*)d_in[7];
  const float* sa_Wv = (const float*)d_in[8];  const float* sa_bv = (const float*)d_in[9];
  const float* sa_Wp = (const float*)d_in[10]; const float* sa_bp = (const float*)d_in[11];
  const float* ca_Wq = (const float*)d_in[12]; const float* ca_bq = (const float*)d_in[13];
  const float* ca_Wk = (const float*)d_in[14]; const float* ca_bk = (const float*)d_in[15];
  const float* ca_Wv = (const float*)d_in[16]; const float* ca_bv = (const float*)d_in[17];
  const float* ca_Wp = (const float*)d_in[18]; const float* ca_bp = (const float*)d_in[19];
  const float* mlp_W1 = (const float*)d_in[20]; const float* mlp_b1 = (const float*)d_in[21];
  const float* mlp_W2 = (const float*)d_in[22]; const float* mlp_b2 = (const float*)d_in[23];
  const float* ln1_g = (const float*)d_in[24]; const float* ln1_b = (const float*)d_in[25];
  const float* ln2_g = (const float*)d_in[26]; const float* ln2_b = (const float*)d_in[27];
  const float* ln3_g = (const float*)d_in[28]; const float* ln3_b = (const float*)d_in[29];

  const size_t BND = (size_t)CB * CN * CD;       // 4M elems
  char* ws = (char*)d_ws;
  size_t off = 0;
  auto alloc = [&](size_t bytes) -> char* {
    char* p = ws + off;
    off += (bytes + 255) & ~(size_t)255;
    return p;
  };

  // bf16 transposed weights
  __bf16* wt_saq = (__bf16*)alloc((size_t)CD * CD * 2);
  __bf16* wt_sak = (__bf16*)alloc((size_t)CD * CD * 2);
  __bf16* wt_sav = (__bf16*)alloc((size_t)CD * CD * 2);
  __bf16* wt_sap = (__bf16*)alloc((size_t)CD * CD * 2);
  __bf16* wt_caq = (__bf16*)alloc((size_t)CD * CD * 2);
  __bf16* wt_cak = (__bf16*)alloc((size_t)CD * CD * 2);
  __bf16* wt_cav = (__bf16*)alloc((size_t)CD * CD * 2);
  __bf16* wt_cap = (__bf16*)alloc((size_t)CD * CD * 2);
  __bf16* wt_w1  = (__bf16*)alloc((size_t)CD * CDM * 2);
  __bf16* wt_w2  = (__bf16*)alloc((size_t)CDM * CD * 2);

  // fp32 buffers
  float* resid = (float*)alloc(BND * 4);
  float* x2    = (float*)alloc(BND * 4);
  float* qk    = (float*)alloc(BND * 4);
  float* big   = (float*)alloc(BND * 4 * 4);  // q|k|v|attno, later reused as h
  float* qbuf  = big;
  float* kbuf  = big + BND;
  float* vbuf  = big + 2 * BND;
  float* attno = big + 3 * BND;
  float* kin   = attno;   // alias: kin dead before attno is written
  float* hbuf  = big;     // alias: q/k/v/attno dead when MLP runs

  const int M = CB * CN;  // 4096
  dim3 tb(256);
  dim3 gD(CD / 128, M / 128);    // N=1024 GEMMs
  dim3 gDM(CDM / 128, M / 128);  // N=4096 GEMM
  dim3 ga(CN / 128, CB * CH);    // attention

  // weight conversion
  wconv_kernel<<<dim3(CD / 32, CD / 32), dim3(32, 8), 0, stream>>>(sa_Wq, wt_saq, CD, CD);
  wconv_kernel<<<dim3(CD / 32, CD / 32), dim3(32, 8), 0, stream>>>(sa_Wk, wt_sak, CD, CD);
  wconv_kernel<<<dim3(CD / 32, CD / 32), dim3(32, 8), 0, stream>>>(sa_Wv, wt_sav, CD, CD);
  wconv_kernel<<<dim3(CD / 32, CD / 32), dim3(32, 8), 0, stream>>>(sa_Wp, wt_sap, CD, CD);
  wconv_kernel<<<dim3(CD / 32, CD / 32), dim3(32, 8), 0, stream>>>(ca_Wq, wt_caq, CD, CD);
  wconv_kernel<<<dim3(CD / 32, CD / 32), dim3(32, 8), 0, stream>>>(ca_Wk, wt_cak, CD, CD);
  wconv_kernel<<<dim3(CD / 32, CD / 32), dim3(32, 8), 0, stream>>>(ca_Wv, wt_cav, CD, CD);
  wconv_kernel<<<dim3(CD / 32, CD / 32), dim3(32, 8), 0, stream>>>(ca_Wp, wt_cap, CD, CD);
  wconv_kernel<<<dim3(CDM / 32, CD / 32), dim3(32, 8), 0, stream>>>(mlp_W1, wt_w1, CD, CDM);
  wconv_kernel<<<dim3(CD / 32, CDM / 32), dim3(32, 8), 0, stream>>>(mlp_W2, wt_w2, CDM, CD);

  int n4 = (int)(BND / 4);
  copy_kernel<<<n4 / 256, tb, 0, stream>>>(tokens_dec, resid, n4);

  // ---- self attention ----
  ln_kernel<<<M, tb, 0, stream>>>(resid, ln1_g, ln1_b, PE_dec, x2, qk);
  gemm_kernel<<<gD, tb, 0, stream>>>(qk, wt_saq, sa_bq, nullptr, qbuf, M, CD, CD, 0);
  gemm_kernel<<<gD, tb, 0, stream>>>(qk, wt_sak, sa_bk, nullptr, kbuf, M, CD, CD, 0);
  gemm_kernel<<<gD, tb, 0, stream>>>(x2, wt_sav, sa_bv, nullptr, vbuf, M, CD, CD, 0);
  attn_kernel<<<ga, tb, 0, stream>>>(qbuf, kbuf, vbuf, attno);
  gemm_kernel<<<gD, tb, 0, stream>>>(attno, wt_sap, sa_bp, resid, resid, M, CD, CD, 2);

  // ---- cross attention ----
  ln_kernel<<<M, tb, 0, stream>>>(resid, ln2_g, ln2_b, PE_dec, x2, qk);
  addpe_kernel<<<n4 / 256, tb, 0, stream>>>(tokens_enc, PE_enc, kin, n4);
  gemm_kernel<<<gD, tb, 0, stream>>>(qk, wt_caq, ca_bq, nullptr, qbuf, M, CD, CD, 0);
  gemm_kernel<<<gD, tb, 0, stream>>>(kin, wt_cak, ca_bk, nullptr, kbuf, M, CD, CD, 0);
  gemm_kernel<<<gD, tb, 0, stream>>>(tokens_enc, wt_cav, ca_bv, nullptr, vbuf, M, CD, CD, 0);
  attn_kernel<<<ga, tb, 0, stream>>>(qbuf, kbuf, vbuf, attno);
  gemm_kernel<<<gD, tb, 0, stream>>>(attno, wt_cap, ca_bp, resid, resid, M, CD, CD, 2);

  // ---- MLP ----
  ln_kernel<<<M, tb, 0, stream>>>(resid, ln3_g, ln3_b, nullptr, x2, nullptr);
  gemm_kernel<<<gDM, tb, 0, stream>>>(x2, wt_w1, mlp_b1, nullptr, hbuf, M, CDM, CD, 1);
  gemm_kernel<<<gD, tb, 0, stream>>>(hbuf, wt_w2, mlp_b2, resid, (float*)d_out, M, CD, CDM, 2);
}
